// RNN_19868518711900
// MI455X (gfx1250) — matrix operand order repaired and checked by reading, not compiled, not run on hardware
//
#include <hip/hip_runtime.h>
#include <hip/hip_bf16.h>

// ---------------------------------------------------------------------------
// Persistent WMMA RNN scan for MI455X (gfx1250, wave32).
//   * 4 workgroups, one per 16-row batch tile (recurrence is batch-parallel).
//   * 512 threads = 16 waves; each wave owns two 16x16 output tiles of
//     tmp_h (waves 0-7) or tmp_r (waves 8-15).
//   * v_wmma_f32_16x16x32_f16, f32 accumulation; K-loop software-pipelined
//     (distance-3 global B prefetch, distance-1 LDS A prefetch, unroll 4).
//   * Ping-pong f16 A-panels (x|pad|h|r) in LDS -> ONE barrier per step.
//   * f32 reservoir copy in LDS for the 0.9r + 0.1*tmp_r update.
//   * Weights packed once to f16 in d_ws, streamed from L2 (~540KB << 192MB).
//   * Head weights resident in VGPRs; noise loaded before the K-loop and
//     next step's noise lines prefetched (global_prefetch_b8).
// ---------------------------------------------------------------------------

typedef _Float16 v8h  __attribute__((ext_vector_type(8)));
typedef _Float16 v16h __attribute__((ext_vector_type(16)));
typedef float    v8f  __attribute__((ext_vector_type(8)));

#define N_HID 256
#define N_RES 256
#define N_IN  16
#define N_OUT 2
#define BATCH 64
#define SEQ   2048

#define KH    544            // x(16) + pad(16) + h(256) + r(256)
#define KR    512            // h(256) + r(256)
#define AST   552            // LDS A-panel row stride in halves (544 + 8 pad)

// d_ws layout (bytes), all chunks 32B-aligned
#define OFF_WH   0                          // f16 [256][544]
#define OFF_WR   (OFF_WH + 256*KH*2)        // f16 [256][512]
#define OFF_WO   (OFF_WR + 256*KR*2)        // f16 [16][256]  (W_out padded to 16 rows)
#define OFF_WP   (OFF_WO + 16*256*2)        // f16 [16][256]  (W_pr)
#define OFF_BH   (OFF_WP + 16*256*2)        // f32 [256]  b_in+b_hh+b_sf
#define OFF_BR   (OFF_BH + 256*4)           // f32 [256]  b_fs+b_res
#define OFF_BO   (OFF_BR + 256*4)           // f32 [16]   b_out padded
#define OFF_BP   (OFF_BO + 64)              // f32 [16]   b_pr

__global__ __launch_bounds__(256) void pack_weights_f16(
    const float* __restrict__ Win,  const float* __restrict__ bin,
    const float* __restrict__ Whh,  const float* __restrict__ bhh,
    const float* __restrict__ Wsf,  const float* __restrict__ bsf,
    const float* __restrict__ Wfs,  const float* __restrict__ bfs,
    const float* __restrict__ Wres, const float* __restrict__ bres,
    const float* __restrict__ Wout, const float* __restrict__ bout,
    const float* __restrict__ Wpr,  const float* __restrict__ bpr,
    char* __restrict__ ws)
{
    _Float16* wh = (_Float16*)(ws + OFF_WH);
    _Float16* wr = (_Float16*)(ws + OFF_WR);
    _Float16* wo = (_Float16*)(ws + OFF_WO);
    _Float16* wp = (_Float16*)(ws + OFF_WP);
    float* bh = (float*)(ws + OFF_BH);
    float* br = (float*)(ws + OFF_BR);
    float* bo = (float*)(ws + OFF_BO);
    float* bp = (float*)(ws + OFF_BP);

    const int tid  = blockIdx.x * blockDim.x + threadIdx.x;
    const int nthr = gridDim.x * blockDim.x;

    for (int i = tid; i < 256 * KH; i += nthr) {
        int row = i / KH, c = i % KH;
        float v;
        if      (c < 16)  v = Win[row * 16 + c];
        else if (c < 32)  v = 0.f;
        else if (c < 288) v = Whh[row * 256 + (c - 32)];
        else              v = Wsf[row * 256 + (c - 288)];
        wh[i] = (_Float16)v;
    }
    for (int i = tid; i < 256 * KR; i += nthr) {
        int row = i / KR, c = i % KR;
        float v = (c < 256) ? Wfs[row * 256 + c] : Wres[row * 256 + (c - 256)];
        wr[i] = (_Float16)v;
    }
    for (int i = tid; i < 16 * 256; i += nthr) {
        int row = i >> 8, c = i & 255;
        wo[i] = (_Float16)((row < N_OUT) ? Wout[row * 256 + c] : 0.f);
        wp[i] = (_Float16)Wpr[row * 256 + c];
    }
    for (int i = tid; i < 256; i += nthr) {
        bh[i] = bin[i] + bhh[i] + bsf[i];
        br[i] = bfs[i] + bres[i];
    }
    if (tid < 16) {
        bo[tid] = (tid < N_OUT) ? bout[tid] : 0.f;
        bp[tid] = bpr[tid];
    }
}

__device__ __forceinline__ v16h load_a_frag(const _Float16* ap) {
    v8h lo = *(const v8h*)(ap);        // K = ksel*16 .. +7
    v8h hi = *(const v8h*)(ap + 8);    // K = ksel*16 + 8 .. +15
    return __builtin_shufflevector(lo, hi, 0,1,2,3,4,5,6,7,8,9,10,11,12,13,14,15);
}

// Software-pipelined dual-tile GEMM: distance-3 rotating prefetch on the
// global B fragments, distance-1 on the LDS A fragment. Tail prefetches read
// slightly past the logical matrix but stay inside d_ws / the LDS allocation.
template <int NK>
__device__ __forceinline__ void gemm_pair(const _Float16* __restrict__ ap,
                                          const _Float16* __restrict__ b0p,
                                          const _Float16* __restrict__ b1p,
                                          v8f& acc0, v8f& acc1)
{
    v16h a = load_a_frag(ap);
    v16h w0[3], w1[3];
#pragma unroll
    for (int s = 0; s < 3; ++s) {
        w0[s] = *(const v16h*)(b0p + s * 32);
        w1[s] = *(const v16h*)(b1p + s * 32);
    }
#pragma unroll 4
    for (int kc = 0; kc < NK; ++kc) {
        v16h ca = a, c0 = w0[0], c1 = w1[0];
        w0[0] = w0[1]; w0[1] = w0[2];
        w1[0] = w1[1]; w1[1] = w1[2];
        w0[2] = *(const v16h*)(b0p + (kc + 3) * 32);
        w1[2] = *(const v16h*)(b1p + (kc + 3) * 32);
        a = load_a_frag(ap + (kc + 1) * 32);
        acc0 = __builtin_amdgcn_wmma_f32_16x16x32_f16(false, ca, false, c0, (short)0, acc0, false, false);
        acc1 = __builtin_amdgcn_wmma_f32_16x16x32_f16(false, ca, false, c1, (short)0, acc1, false, false);
    }
}

__global__ __launch_bounds__(512, 1) void rnn_persistent(
    const float* __restrict__ xin,   const float* __restrict__ h0,
    const float* __restrict__ r0,    const float* __restrict__ noise,
    const char*  __restrict__ ws,
    float* __restrict__ hs, float* __restrict__ outs, float* __restrict__ priors)
{
    const _Float16* WH = (const _Float16*)(ws + OFF_WH);
    const _Float16* WR = (const _Float16*)(ws + OFF_WR);
    const _Float16* WO = (const _Float16*)(ws + OFF_WO);
    const _Float16* WP = (const _Float16*)(ws + OFF_WP);
    const float* BH = (const float*)(ws + OFF_BH);
    const float* BR = (const float*)(ws + OFF_BR);
    const float* BO = (const float*)(ws + OFF_BO);
    const float* BP = (const float*)(ws + OFF_BP);

    // Ping-pong f16 A-panels: step t reads panel (t&1), writes panel (t+1)&1.
    __shared__ __align__(32) _Float16 Apan[2][16 * AST];
    __shared__ __align__(16) float    R32[16 * 256]; // f32 reservoir state

    const int tid  = threadIdx.x;
    const int wave = tid >> 5;
    const int lane = tid & 31;
    const int row  = lane & 15;       // A row / B column / C column index
    const int ksel = lane >> 4;       // K-half select per ISA lane layout
    const int b0   = blockIdx.x * 16; // batch tile origin

    // ---- prologue: seed h, r, x(t=0) into panel 0; zero-pad both panels
    for (int i = tid; i < 16 * 256; i += 512) {
        int m = i >> 8, c = i & 255;
        float hv = h0[(size_t)(b0 + m) * 256 + c];
        float rv = r0[(size_t)(b0 + m) * 256 + c];
        Apan[0][m * AST + 32  + c] = (_Float16)hv;
        Apan[0][m * AST + 288 + c] = (_Float16)rv;
        R32[i] = rv;
    }
    for (int i = tid; i < 256; i += 512) {
        int m = i >> 4, c = i & 15;
        Apan[0][m * AST + 16 + c] = (_Float16)0.f;
        Apan[1][m * AST + 16 + c] = (_Float16)0.f;
        Apan[0][m * AST + c] = (_Float16)xin[((size_t)(b0 + m) * SEQ) * N_IN + c];
    }

    const bool isH = (wave < 8);
    const int  nb0 = (isH ? wave : wave - 8) * 32;  // first 16-col tile base
    const int  nb1 = nb0 + 16;
    const _Float16* Wcat = isH ? WH : WR;
    const int  Kdim = isH ? KH : KR;

    const _Float16* bRow0 = Wcat + (size_t)(nb0 + row) * Kdim + ksel * 16;
    const _Float16* bRow1 = bRow0 + (size_t)16 * Kdim;

    // ---- resident head weights (waves 0/1): 8 fragments each, kept in VGPRs
    v16h headW[8];
    if (wave < 2) {
        const _Float16* hw = (wave == 0 ? WO : WP) + row * 256 + ksel * 16;
#pragma unroll
        for (int kc = 0; kc < 8; ++kc) headW[kc] = *(const v16h*)(hw + kc * 32);
    }
    const float headB = (wave == 0) ? BO[row] : BP[row];

    __syncthreads();

    for (int t = 0; t < SEQ; ++t) {
        const _Float16* Ar = Apan[t & 1];            // state/x for step t
        _Float16*       Aw = Apan[(t + 1) & 1];      // state/x for step t+1

        // ---- issue this step's noise loads first: latency hides under K-loop
        float nz[16];
        if (isH) {
#pragma unroll
            for (int j = 0; j < 2; ++j)
#pragma unroll
                for (int g = 0; g < 8; ++g) {
                    int m   = ksel * 8 + g;
                    int col = (j ? nb1 : nb0) + row;
                    nz[j * 8 + g] =
                        noise[((size_t)t * BATCH + (b0 + m)) * N_HID + col];
                }
        }

        // ---- phase 1: two 16x16 tiles per wave via f16 WMMA, f32 accum
        v8f acc0 = {}, acc1 = {};
        if (isH) gemm_pair<17>(&Ar[row * AST + ksel * 16],      bRow0, bRow1, acc0, acc1);
        else     gemm_pair<16>(&Ar[row * AST + 32 + ksel * 16], bRow0, bRow1, acc0, acc1);

        // ---- phase 2: state update into the OTHER panel (no barrier needed:
        //      nobody reads Aw until after the end-of-step barrier)
        if (isH) {
#pragma unroll
            for (int j = 0; j < 2; ++j) {
                const int   col = (j ? nb1 : nb0) + row;
                const float bia = BH[col];
                const v8f&  acc = j ? acc1 : acc0;
#pragma unroll
                for (int g = 0; g < 8; ++g) {
                    int   m  = ksel * 8 + g;
                    float v  = acc[g] + bia;       v = v > 0.f ? v : 0.f;
                    float hn = v + nz[j * 8 + g];
                    hs[((size_t)(b0 + m) * SEQ + t) * N_HID + col] = hn;
                    Aw[m * AST + 32 + col] = (_Float16)hn;
                }
            }
            if (t + 1 < SEQ) {                       // warm next step's noise
#pragma unroll
                for (int j = 0; j < 2; ++j)
#pragma unroll
                    for (int g = 0; g < 8; ++g) {
                        int m   = ksel * 8 + g;
                        int col = (j ? nb1 : nb0) + row;
                        __builtin_prefetch(
                            &noise[((size_t)(t + 1) * BATCH + (b0 + m)) * N_HID + col], 0, 3);
                    }
            }
        } else {
#pragma unroll
            for (int j = 0; j < 2; ++j) {
                const int   col = (j ? nb1 : nb0) + row;
                const float bia = BR[col];
                const v8f&  acc = j ? acc1 : acc0;
#pragma unroll
                for (int g = 0; g < 8; ++g) {
                    int   m  = ksel * 8 + g;
                    float tv = acc[g] + bia;       tv = tv > 0.f ? tv : 0.f;
                    float rn = 0.9f * R32[m * 256 + col] + 0.1f * tv;
                    R32[m * 256 + col] = rn;
                    Aw[m * AST + 288 + col] = (_Float16)rn;
                }
            }
            if (wave == 15 && t + 1 < SEQ) {         // stage x(t+1)
                for (int i = lane; i < 256; i += 32) {
                    int m = i >> 4, c = i & 15;
                    Aw[m * AST + c] =
                        (_Float16)xin[((size_t)(b0 + m) * SEQ + (t + 1)) * N_IN + c];
                }
            }
        }
        __syncthreads();   // single barrier per step: new panel now visible

        // ---- phase 3: tiny heads on the new state (waves 0 and 1).
        //      Aw is read-only for everyone during step t+1 (it writes the
        //      other panel), so heads may lag into the next step safely.
        if (wave == 0) {                            // out = clip(h W_out^T + b)
            v8f acc = {};
            const _Float16* ah = &Aw[row * AST + 32 + ksel * 16];
#pragma unroll
            for (int kc = 0; kc < 8; ++kc) {
                v16h a = load_a_frag(ah + kc * 32);
                acc = __builtin_amdgcn_wmma_f32_16x16x32_f16(false, a, false, headW[kc], (short)0, acc, false, false);
            }
            if (row < N_OUT) {
#pragma unroll
                for (int g = 0; g < 8; ++g) {
                    int   m = ksel * 8 + g;
                    float v = acc[g] + headB;
                    v = fminf(fmaxf(v, -2.f), 2.f);
                    outs[((size_t)(b0 + m) * SEQ + t) * N_OUT + row] = v;
                }
            }
        } else if (wave == 1) {                     // prior = sigmoid(r W_pr^T + b)
            v8f acc = {};
            const _Float16* ah = &Aw[row * AST + 288 + ksel * 16];
#pragma unroll
            for (int kc = 0; kc < 8; ++kc) {
                v16h a = load_a_frag(ah + kc * 32);
                acc = __builtin_amdgcn_wmma_f32_16x16x32_f16(false, a, false, headW[kc], (short)0, acc, false, false);
            }
#pragma unroll
            for (int g = 0; g < 8; ++g) {
                int   m = ksel * 8 + g;
                float v = acc[g] + headB;
                v = 1.f / (1.f + expf(-v));
                priors[((size_t)(b0 + m) * SEQ + t) * N_IN + row] = v;
            }
        }
    }
}

extern "C" void kernel_launch(void* const* d_in, const int* in_sizes, int n_in,
                              void* d_out, int out_size, void* d_ws, size_t ws_size,
                              hipStream_t stream) {
    (void)in_sizes; (void)n_in; (void)out_size; (void)ws_size;
    const float* xin   = (const float*)d_in[0];
    const float* h0    = (const float*)d_in[1];
    const float* r0    = (const float*)d_in[2];
    const float* noise = (const float*)d_in[3];
    const float* Win   = (const float*)d_in[4];
    const float* bin   = (const float*)d_in[5];
    const float* Whh   = (const float*)d_in[6];
    const float* bhh   = (const float*)d_in[7];
    const float* Wsf   = (const float*)d_in[8];
    const float* bsf   = (const float*)d_in[9];
    const float* Wfs   = (const float*)d_in[10];
    const float* bfs   = (const float*)d_in[11];
    const float* Wres  = (const float*)d_in[12];
    const float* bres  = (const float*)d_in[13];
    const float* Wout  = (const float*)d_in[14];
    const float* bout  = (const float*)d_in[15];
    const float* Wpr   = (const float*)d_in[16];
    const float* bpr   = (const float*)d_in[17];

    float* hs     = (float*)d_out;
    float* outs   = hs   + (size_t)BATCH * SEQ * N_HID;
    float* priors = outs + (size_t)BATCH * SEQ * N_OUT;
    char*  ws     = (char*)d_ws;

    pack_weights_f16<<<128, 256, 0, stream>>>(Win, bin, Whh, bhh, Wsf, bsf,
                                              Wfs, bfs, Wres, bres,
                                              Wout, bout, Wpr, bpr, ws);
    rnn_persistent<<<4, 512, 0, stream>>>(xin, h0, r0, noise, ws, hs, outs, priors);
}
